// SelfAttentionLayer_23295902613523
// MI455X (gfx1250) — matrix operand order repaired and checked by reading, not compiled, not run on hardware
//
#include <hip/hip_runtime.h>
#include <hip/hip_bf16.h>
#include <math.h>

// ---------------- problem constants ----------------
#define BATCH 8
#define SEQ   16384
#define DMODEL 256
#define NHEAD 8
#define HDIM  32
#define ROWS_TOTAL (BATCH * SEQ)            // 131072
#define TILE_ROWS 64                        // rows per block (4 waves x 16)
#define CTX_ROWS 512                        // seq rows per single-wave ctx block

// ---------------- vector types ----------------
typedef __attribute__((ext_vector_type(8)))  __bf16 v8bf;
typedef __attribute__((ext_vector_type(16))) __bf16 v16bf;
typedef __attribute__((ext_vector_type(8)))  float  v8f;
typedef __attribute__((ext_vector_type(4)))  unsigned int u32x4;
typedef __attribute__((ext_vector_type(4)))  int i32x4;
typedef __attribute__((ext_vector_type(8)))  int i32x8;

__device__ __forceinline__ v16bf cat8(v8bf a, v8bf b) {
    return __builtin_shufflevector(a, b, 0,1,2,3,4,5,6,7,8,9,10,11,12,13,14,15);
}

__device__ __forceinline__ unsigned short f2bfu(float f) {
    union { float f; unsigned u; } a; a.f = f;
    unsigned u = a.u;
    unsigned r = u + 0x7FFFu + ((u >> 16) & 1u);   // round-to-nearest-even
    return (unsigned short)(r >> 16);
}

// ---------------- TDM: 2-D tile load  global -> LDS  (D# per ISA 08 Sec.8) ----------------
// data_size = 2 bytes. Tile: tile_d0 (fastest, contiguous) x tile_d1 rows, row stride
// stride0 elements. Generic LDS pointer truncates to the LDS byte offset (aperture rule).
__device__ __forceinline__ void tdm_load_2d_bf16(void* lds_dst, const void* gsrc,
                                                 unsigned tensor_d0, unsigned tensor_d1,
                                                 unsigned tile_d0, unsigned tile_d1,
                                                 unsigned stride0) {
    unsigned long long ga = (unsigned long long)gsrc;
    u32x4 g0;
    g0[0] = 1u;                                            // count=1, user descriptor
    g0[1] = (unsigned)(unsigned long long)lds_dst;         // lds_addr (byte offset)
    g0[2] = (unsigned)(ga & 0xFFFFFFFFu);                  // global_addr[31:0]
    g0[3] = (unsigned)((ga >> 32) & 0x01FFFFFFu) | 0x80000000u; // addr[56:32] | type=2
    i32x8 g1;
    g1[0] = 1 << 16;                                       // data_size = 1 -> 2 bytes
    g1[1] = (int)((tensor_d0 & 0xFFFFu) << 16);            // tensor_dim0[15:0]
    g1[2] = (int)(((tensor_d0 >> 16) & 0xFFFFu) | ((tensor_d1 & 0xFFFFu) << 16));
    g1[3] = (int)(((tensor_d1 >> 16) & 0xFFFFu) | ((tile_d0 & 0xFFFFu) << 16));
    g1[4] = (int)(tile_d1 & 0xFFFFu);                      // tile_dim1 (tile_dim2 = 0)
    g1[5] = (int)stride0;                                  // tensor_dim0_stride[31:0]
    g1[6] = 0;                                             // stride0[47:32], stride1 lo
    g1[7] = 0;                                             // stride1 hi
    i32x4 z4 = {};
#if __clang_major__ >= 23
    i32x8 z8 = {};
    __builtin_amdgcn_tensor_load_to_lds(g0, g1, z4, z4, z8, 0);
#else
    __builtin_amdgcn_tensor_load_to_lds(g0, g1, z4, z4, 0);
#endif
}

// ---------------- kernel 0: weight convert + zero accumulators ----------------
__global__ void prep_kernel(const float* Wq, const float* Wk, const float* Wv, const float* Wo,
                            unsigned short* wq, unsigned short* wk, unsigned short* wv,
                            unsigned short* wo, float* ksum, float* ctx) {
    int i = blockIdx.x * blockDim.x + threadIdx.x;   // 65536 threads
    if (i < DMODEL * DMODEL) {
        wq[i] = f2bfu(Wq[i]);
        wk[i] = f2bfu(Wk[i]);
        wv[i] = f2bfu(Wv[i]);
        wo[i] = f2bfu(Wo[i]);
        ctx[i & (BATCH * NHEAD * HDIM * HDIM - 1)] = 0.f;  // 65536 == B*H*32*32
        if (i < BATCH * NHEAD * HDIM) ksum[i] = 0.f;
    }
}

// ---------------- shared GEMM helper: out[m][n] = act( A(lds) @ W^T + bias ) ----------------
__device__ __forceinline__ void gemm_256(const unsigned short* a_sh, const __bf16* W,
                                         const float* bias, unsigned short* out,
                                         size_t row0, int wave, int lane, int act) {
    v8f acc[16];
    #pragma unroll
    for (int i = 0; i < 16; ++i) { v8f z = {}; acc[i] = z; }

    const int l16  = lane & 15;
    const int half = lane >> 4;
    const __bf16* xs = (const __bf16*)a_sh;

    #pragma unroll
    for (int kk = 0; kk < 8; ++kk) {
        int abase = (wave * 16 + l16) * DMODEL + kk * 32 + half * 8;
        v8bf a0 = *(const v8bf*)(xs + abase);
        v8bf a1 = *(const v8bf*)(xs + abase + 16);
        v16bf af = cat8(a0, a1);
        #pragma unroll
        for (int nt = 0; nt < 16; ++nt) {
            const __bf16* wp = W + (nt * 16 + l16) * DMODEL + kk * 32 + half * 8;
            v8bf b0 = *(const v8bf*)wp;
            v8bf b1 = *(const v8bf*)(wp + 16);
            acc[nt] = __builtin_amdgcn_wmma_f32_16x16x32_bf16(
                false, af, false, cat8(b0, b1), (short)0, acc[nt], false, false);
        }
    }
    #pragma unroll
    for (int nt = 0; nt < 16; ++nt) {
        int col = nt * 16 + l16;
        float b = bias[col];
        #pragma unroll
        for (int r = 0; r < 8; ++r) {
            float v = acc[nt][r] + b;
            if (act) v = (v > 0.f) ? (v + 1.f) : __expf(v);   // elu(v)+1
            int rl = wave * 16 + r + half * 8;
            out[(row0 + rl) * DMODEL + col] = f2bfu(v);
        }
    }
}

// ---------------- kernel 1: x = tgt+pos -> bf16 LDS -> Q,K,V GEMMs ----------------
__global__ void __launch_bounds__(128)
qkv_kernel(const float* tgt, const float* pos,
           const unsigned short* wq, const unsigned short* wk, const unsigned short* wv,
           const float* bq, const float* bk, const float* bv,
           unsigned short* qb, unsigned short* kb, unsigned short* vb) {
    __shared__ unsigned short xt[TILE_ROWS * DMODEL];   // 32 KB
    const int t = threadIdx.x;                          // 128 threads = 4 waves
    const size_t row0 = (size_t)blockIdx.x * TILE_ROWS;

    if (t == 0) { __builtin_prefetch(wq, 0, 0); __builtin_prefetch(wk, 0, 0); __builtin_prefetch(wv, 0, 0); }

    const float4* tg4 = (const float4*)(tgt + row0 * DMODEL);
    const float4* pp4 = (const float4*)(pos + row0 * DMODEL);
    #pragma unroll 4
    for (int i = 0; i < (TILE_ROWS * DMODEL) / (128 * 4); ++i) {
        int e = i * 128 + t;
        float4 a = tg4[e], p = pp4[e];
        int base = e * 4;
        xt[base + 0] = f2bfu(a.x + p.x);
        xt[base + 1] = f2bfu(a.y + p.y);
        xt[base + 2] = f2bfu(a.z + p.z);
        xt[base + 3] = f2bfu(a.w + p.w);
    }
    __syncthreads();

    const int lane = t & 31, wave = t >> 5;
    gemm_256(xt, (const __bf16*)wq, bq, qb, row0, wave, lane, 1);
    gemm_256(xt, (const __bf16*)wk, bk, kb, row0, wave, lane, 1);
    gemm_256(xt, (const __bf16*)wv, bv, vb, row0, wave, lane, 0);
}

// ---------------- kernel 2: context[b,h] += K^T V, TDM-staged tiles, 1 wave/block ----------
// Block = 32 threads (one wave) so TDM descriptors are blockIdx-uniform (SGPR-clean).
// Double-buffered 32x32 bf16 k/v tiles moved by tensor_load_to_lds, synced with TENSORcnt.
__global__ void __launch_bounds__(32)
ctx_kernel(const unsigned short* kbuf, const unsigned short* vbuf,
           float* ctx, float* ksum) {
    __shared__ unsigned short kts[2][HDIM * HDIM];      // 2 x 2 KB
    __shared__ unsigned short vts[2][HDIM * HDIM];      // 2 x 2 KB
    const int bid   = blockIdx.x;                       // B*H*(SEQ/CTX_ROWS) = 2048
    const int chunk = bid & 31;
    const int h     = (bid >> 5) & (NHEAD - 1);
    const int b     = bid >> 8;
    const int n0    = chunk * CTX_ROWS;

    const int lane = threadIdx.x;
    const int l16 = lane & 15, half = lane >> 4;

    const unsigned short* kg = kbuf + (size_t)b * SEQ * DMODEL + (size_t)n0 * DMODEL + h * HDIM;
    const unsigned short* vg = vbuf + (size_t)b * SEQ * DMODEL + (size_t)n0 * DMODEL + h * HDIM;

    // preload iteration 0 tiles (32 rows x 32 cols, row stride DMODEL elements)
    tdm_load_2d_bf16(kts[0], kg, HDIM, CTX_ROWS, HDIM, 32, DMODEL);
    tdm_load_2d_bf16(vts[0], vg, HDIM, CTX_ROWS, HDIM, 32, DMODEL);

    v8f acc[2][2];
    #pragma unroll
    for (int i = 0; i < 2; ++i)
        #pragma unroll
        for (int j = 0; j < 2; ++j) { v8f z = {}; acc[i][j] = z; }

    float ks_lane = 0.f;                                // k_sum for column = lane

    for (int it = 0; it < CTX_ROWS / 32; ++it) {        // 16 iterations
        // issue next tiles (wraps harmlessly on last iteration to keep wait imm constant)
        int nxt = (it + 1) & 15;
        tdm_load_2d_bf16(kts[(it + 1) & 1], kg + (size_t)nxt * 32 * DMODEL,
                         HDIM, CTX_ROWS, HDIM, 32, DMODEL);
        tdm_load_2d_bf16(vts[(it + 1) & 1], vg + (size_t)nxt * 32 * DMODEL,
                         HDIM, CTX_ROWS, HDIM, 32, DMODEL);
        // wait until only the 2 just-issued loads are outstanding -> current tiles ready
        __builtin_amdgcn_s_wait_tensorcnt(2);

        const __bf16* kt = (const __bf16*)kts[it & 1];
        const __bf16* vt = (const __bf16*)vts[it & 1];

        v16bf af[2], bfr[2];
        #pragma unroll
        for (int dt = 0; dt < 2; ++dt) {                // A = K^T tile (d x n) from LDS
            v16bf a;
            int d = dt * 16 + l16;
            #pragma unroll
            for (int i = 0; i < 8; ++i) {
                a[i]     = kt[(half * 16 + i) * HDIM + d];
                a[8 + i] = kt[(half * 16 + 8 + i) * HDIM + d];
            }
            af[dt] = a;
        }
        #pragma unroll
        for (int vti = 0; vti < 2; ++vti) {             // B = V tile (n x v) from LDS
            v16bf bb;
            int vc = vti * 16 + l16;
            #pragma unroll
            for (int i = 0; i < 16; ++i)
                bb[i] = vt[(half * 16 + i) * HDIM + vc];
            bfr[vti] = bb;
        }
        #pragma unroll
        for (int dt = 0; dt < 2; ++dt)
            #pragma unroll
            for (int vti = 0; vti < 2; ++vti)
                acc[dt][vti] = __builtin_amdgcn_wmma_f32_16x16x32_bf16(
                    false, af[dt], false, bfr[vti], (short)0, acc[dt][vti], false, false);

        // k_sum from the LDS-resident k tile (data already moved by TDM)
        #pragma unroll
        for (int n = 0; n < 32; ++n) ks_lane += (float)kt[n * HDIM + lane];
    }

    float* cp = ctx + ((size_t)b * NHEAD + h) * HDIM * HDIM;
    #pragma unroll
    for (int dt = 0; dt < 2; ++dt)
        #pragma unroll
        for (int vti = 0; vti < 2; ++vti)
            #pragma unroll
            for (int r = 0; r < 8; ++r) {
                int d  = dt * 16 + r + half * 8;
                int vv = vti * 16 + l16;
                atomicAdd(&cp[d * HDIM + vv], acc[dt][vti][r]);
            }
    atomicAdd(&ksum[(b * NHEAD + h) * HDIM + lane], ks_lane);
}

// ---------------- kernel 3: context f32 -> bf16 ----------------
__global__ void ctx_cvt_kernel(const float* ctx, unsigned short* ctxb) {
    int i = blockIdx.x * blockDim.x + threadIdx.x;      // 65536
    ctxb[i] = f2bfu(ctx[i]);
}

// ---------------- kernel 4: attn = D_inv*(q@ctx) -> Wo GEMM -> DyT epilogue ----------------
__global__ void __launch_bounds__(128)
out_kernel(const unsigned short* qb, const unsigned short* ctxb, const float* ksum,
           const unsigned short* wo, const float* bo, const float* tgt,
           const float* alpha, const float* rvec, const float* beta, float* out) {
    __shared__ unsigned short at[TILE_ROWS * DMODEL];   // 32 KB
    __shared__ float dinv[TILE_ROWS * NHEAD];           // 2 KB
    const int t = threadIdx.x, lane = t & 31, wave = t >> 5;
    const int l16 = lane & 15, half = lane >> 4;
    const size_t row0 = (size_t)blockIdx.x * TILE_ROWS;
    const size_t b = row0 / SEQ;
    const __bf16* qp = (const __bf16*)qb;

    for (int i = t; i < TILE_ROWS * NHEAD; i += 128) {
        int r = i >> 3, h = i & 7;
        const __bf16* qr = qp + (row0 + r) * DMODEL + h * HDIM;
        const float*  ks = ksum + (b * NHEAD + h) * HDIM;
        float s = 0.f;
        #pragma unroll
        for (int d = 0; d < HDIM; ++d) s += (float)qr[d] * ks[d];
        dinv[i] = 1.0f / (s + 1e-6f);
    }
    __syncthreads();

    const __bf16* cp = (const __bf16*)ctxb + b * NHEAD * HDIM * HDIM;
    for (int ct = 0; ct < 16; ++ct) {
        int h = ct >> 1, vhalf = ct & 1;
        const __bf16* qr = qp + (row0 + wave * 16 + l16) * DMODEL + h * HDIM + half * 16;
        v8bf a0 = *(const v8bf*)qr;
        v8bf a1 = *(const v8bf*)(qr + 8);
        v16bf af = cat8(a0, a1);
        v16bf bf;
        const __bf16* cb = cp + h * HDIM * HDIM + vhalf * 16 + l16;
        #pragma unroll
        for (int i = 0; i < 16; ++i) bf[i] = cb[(half * 16 + i) * HDIM];
        v8f acc = {};
        acc = __builtin_amdgcn_wmma_f32_16x16x32_bf16(false, af, false, bf,
                                                      (short)0, acc, false, false);
        #pragma unroll
        for (int r = 0; r < 8; ++r) {
            int rl = wave * 16 + r + half * 8;
            at[rl * DMODEL + ct * 16 + l16] = f2bfu(acc[r] * dinv[rl * NHEAD + h]);
        }
    }
    __syncthreads();

    v8f acc[16];
    #pragma unroll
    for (int i = 0; i < 16; ++i) { v8f z = {}; acc[i] = z; }
    const __bf16* xs = (const __bf16*)at;
    const __bf16* W  = (const __bf16*)wo;
    #pragma unroll
    for (int kk = 0; kk < 8; ++kk) {
        int abase = (wave * 16 + l16) * DMODEL + kk * 32 + half * 8;
        v8bf a0 = *(const v8bf*)(xs + abase);
        v8bf a1 = *(const v8bf*)(xs + abase + 16);
        v16bf af = cat8(a0, a1);
        #pragma unroll
        for (int nt = 0; nt < 16; ++nt) {
            const __bf16* wp = W + (nt * 16 + l16) * DMODEL + kk * 32 + half * 8;
            v8bf b0 = *(const v8bf*)wp;
            v8bf b1 = *(const v8bf*)(wp + 16);
            acc[nt] = __builtin_amdgcn_wmma_f32_16x16x32_bf16(
                false, af, false, cat8(b0, b1), (short)0, acc[nt], false, false);
        }
    }
    const float al = alpha[0];
    #pragma unroll
    for (int nt = 0; nt < 16; ++nt) {
        int col = nt * 16 + l16;
        float bias = bo[col], rc = rvec[col], bc = beta[col];
        #pragma unroll
        for (int r = 0; r < 8; ++r) {
            int rl = wave * 16 + r + half * 8;
            size_t g = (row0 + rl) * DMODEL + col;
            float tt = tgt[g] + acc[nt][r] + bias;      // residual
            float y  = tanhf(al * tt);                  // DyT
            out[g] = rc * y + bc;
        }
    }
}

// ---------------- launcher ----------------
extern "C" void kernel_launch(void* const* d_in, const int* in_sizes, int n_in,
                              void* d_out, int out_size, void* d_ws, size_t ws_size,
                              hipStream_t stream) {
    const float* tgt = (const float*)d_in[0];
    const float* pos = (const float*)d_in[1];
    const float* Wq  = (const float*)d_in[2];
    const float* bq  = (const float*)d_in[3];
    const float* Wk  = (const float*)d_in[4];
    const float* bk  = (const float*)d_in[5];
    const float* Wv  = (const float*)d_in[6];
    const float* bv  = (const float*)d_in[7];
    const float* Wo  = (const float*)d_in[8];
    const float* bo  = (const float*)d_in[9];
    const float* alpha = (const float*)d_in[10];
    const float* rvec  = (const float*)d_in[11];
    const float* beta  = (const float*)d_in[12];
    float* out = (float*)d_out;

    const size_t QKV_BYTES = (size_t)ROWS_TOTAL * DMODEL * sizeof(unsigned short); // 64 MB
    const size_t W_BYTES   = (size_t)DMODEL * DMODEL * sizeof(unsigned short);     // 128 KB
    char* ws = (char*)d_ws;
    unsigned short* qb  = (unsigned short*)(ws);
    unsigned short* kb  = (unsigned short*)(ws + QKV_BYTES);
    unsigned short* vb  = (unsigned short*)(ws + 2 * QKV_BYTES);
    unsigned short* wqb = (unsigned short*)(ws + 3 * QKV_BYTES);
    unsigned short* wkb = (unsigned short*)(ws + 3 * QKV_BYTES + W_BYTES);
    unsigned short* wvb = (unsigned short*)(ws + 3 * QKV_BYTES + 2 * W_BYTES);
    unsigned short* wob = (unsigned short*)(ws + 3 * QKV_BYTES + 3 * W_BYTES);
    float* ksum = (float*)(ws + 3 * QKV_BYTES + 4 * W_BYTES);
    float* ctx  = (float*)(ws + 3 * QKV_BYTES + 4 * W_BYTES + 16384);
    unsigned short* ctxb = (unsigned short*)(ws + 3 * QKV_BYTES + 4 * W_BYTES + 16384 + 262144);

    prep_kernel<<<DMODEL * DMODEL / 256, 256, 0, stream>>>(Wq, Wk, Wv, Wo,
                                                           wqb, wkb, wvb, wob, ksum, ctx);
    qkv_kernel<<<ROWS_TOTAL / TILE_ROWS, 128, 0, stream>>>(tgt, pos, wqb, wkb, wvb,
                                                           bq, bk, bv, qb, kb, vb);
    ctx_kernel<<<BATCH * NHEAD * (SEQ / CTX_ROWS), 32, 0, stream>>>(kb, vb, ctx, ksum);
    ctx_cvt_kernel<<<BATCH * NHEAD * HDIM * HDIM / 256, 256, 0, stream>>>(ctx, ctxb);
    out_kernel<<<ROWS_TOTAL / TILE_ROWS, 128, 0, stream>>>(qb, ctxb, ksum, wob, bo, tgt,
                                                           alpha, rvec, beta, out);
}
